// Policy_44032004718836
// MI455X (gfx1250) — matrix operand order repaired and checked by reading, not compiled, not run on hardware
//
#include <hip/hip_runtime.h>
#include <hip/hip_bf16.h>
#include <stdint.h>

#define D     512
#define H     8
#define NREQ  16
#define SQ    174          // tokens
#define SP    192          // S padded to K-multiple of 32 for probs@v
#define NACT  17
#define DFF   2048
#define NL    2
#define NB    512          // batch
#define MTOK  (NB*SQ)      // 89088 rows = 128 * 696 exactly
#define DH    64           // head dim

typedef __attribute__((ext_vector_type(16))) __bf16 bf16x16;
typedef __attribute__((ext_vector_type(8)))  float  f32x8;
typedef __attribute__((ext_vector_type(4)))  int    v4i;
typedef unsigned short u16;

__device__ __forceinline__ u16 f2bf(float f) {
  unsigned u = __float_as_uint(f);
  unsigned r = u + 0x7FFFu + ((u >> 16) & 1u);   // RNE
  return (u16)(r >> 16);
}

// ----------------------------------------------------------------
// Async global -> LDS copy of 16 bytes (CDNA5 GLOBAL_LOAD_ASYNC_TO_LDS_B128,
// tracked by ASYNCcnt). Builtin signature (from hipcc diagnostic):
//   (v4i addrspace(1)*, v4i addrspace(3)*, imm offset, imm cpol)
#if defined(__gfx1250__) && __has_builtin(__builtin_amdgcn_global_load_async_to_lds_b128)
#define ASYNC_MODE 2
#elif defined(__gfx1250__)
#define ASYNC_MODE 1
#else
#define ASYNC_MODE 0
#endif

__device__ __forceinline__ void async_cp16(void* lds, const void* g) {
#if ASYNC_MODE == 2
  typedef __attribute__((address_space(1))) v4i* gp_t;
  typedef __attribute__((address_space(3))) v4i* lp_t;
  __builtin_amdgcn_global_load_async_to_lds_b128((gp_t)g, (lp_t)lds, 0, 0);
#elif ASYNC_MODE == 1
  unsigned lo = (unsigned)(uintptr_t)(__attribute__((address_space(3))) void*)lds;
  asm volatile("global_load_async_to_lds_b128 %0, %1, off"
               :: "v"(lo), "v"(g) : "memory");
#else
  *(uint4*)lds = *(const uint4*)g;
#endif
}

__device__ __forceinline__ void wait_async0() {
#if defined(__gfx1250__) && __has_builtin(__builtin_amdgcn_s_wait_asynccnt)
  __builtin_amdgcn_s_wait_asynccnt(0);
#elif defined(__gfx1250__)
  asm volatile("s_wait_asynccnt 0x0" ::: "memory");
#endif
}

// ---------------------------------------------------------------- convert
__global__ void k_cvt(const float* __restrict__ in, u16* __restrict__ out, long long n) {
  long long i = (long long)blockIdx.x * blockDim.x + threadIdx.x;
  long long str = (long long)gridDim.x * blockDim.x;
  for (; i < n; i += str) out[i] = f2bf(in[i]);
}

// ---------------------------------------------------------------- embedding gather
__global__ void k_embed(const float* __restrict__ world, const float* __restrict__ req,
                        const float* __restrict__ veh,
                        const float* __restrict__ Et, const float* __restrict__ Ep,
                        const float* __restrict__ Erid, const float* __restrict__ Erst,
                        const float* __restrict__ Evst, const float* __restrict__ Ecv,
                        float* __restrict__ xf, u16* __restrict__ xb) {
  int tok = blockIdx.x;
  int b = tok / SQ, t = tok % SQ;
  const float* src;
  if (t < 8) {
    const float* w = world + (long long)b * 8;
    if (t == 0)      src = Et  + (long long)(int)roundf(w[0] * 10.f) * D;
    else if (t == 1) src = Ecv + (long long)(int)w[1] * D;
    else if (t < 6)  src = Ep  + (long long)(int)roundf(w[t] * 10.f + 100.f) * D;
    else             src = Et  + (long long)(int)roundf(w[t] * 10.f) * D;
  } else if (t < 8 + NREQ * 10) {
    int r = (t - 8) / 10, j = (t - 8) % 10;
    const float* q = req + ((long long)b * NREQ + r) * 10;
    if (j == 0)      src = Erid + (long long)(int)q[0] * D;
    else if (j < 5)  src = Ep   + (long long)(int)roundf(q[j] * 10.f + 100.f) * D;
    else if (j < 9)  src = Et   + (long long)(int)roundf(q[j] * 10.f) * D;
    else             src = Erst + (long long)(int)q[9] * D;
  } else {
    int j = t - (8 + NREQ * 10);
    const float* v = veh + (long long)b * 6;
    if (j < 2)       src = Ep   + (long long)(int)roundf(v[j] * 10.f + 100.f) * D;
    else if (j == 2) src = Evst + (long long)(int)v[2] * D;
    else             src = Erid + (long long)(int)v[j] * D;
  }
  long long o = (long long)tok * D;
  for (int c = threadIdx.x; c < D; c += blockDim.x) {
    float val = src[c];
    xf[o + c] = val;
    xb[o + c] = f2bf(val);
  }
}

// ================================================================
// FAST GEMM: no bounds checks (M % 128 == 0, N % 128 == 0, K % 32 == 0),
// NT only (B is [N,K] row-major), double-buffered async LDS staging.
// Block 256 thr = 8 waves in 4x2; wave tile 32x64 -> 8 WMMA / k-step.
// ================================================================
struct GemmF {
  const u16* A; int lda;
  const u16* B; int ldb;
  float* Cf; u16* Cb; int ldc;
  int K; const float* bias; int relu;
};

#define BMF 128
#define BNF 128

__global__ void __launch_bounds__(256) k_gemm_fast(GemmF g) {
  __shared__ __align__(16) u16 sA[2][BMF * 32];
  __shared__ __align__(16) u16 sB[2][BNF * 32];
  int tid = threadIdx.x, lane = tid & 31;
  int wv = tid >> 5, wm = wv & 3, wn = wv >> 2;   // 4x2 waves
  int m0 = blockIdx.y * BMF, n0 = blockIdx.x * BNF;

  // staging map: each thread owns two 16B chunks per tile
  int rS = tid >> 2, cS = (tid & 3) * 8;          // row 0..63, chunk in halfs
  const u16* pA0 = g.A + (long long)(m0 + rS) * g.lda + cS;
  const u16* pA1 = pA0 + (long long)64 * g.lda;
  const u16* pB0 = g.B + (long long)(n0 + rS) * g.ldb + cS;
  const u16* pB1 = pB0 + (long long)64 * g.ldb;
  unsigned o0 = rS * 32 + cS, o1 = o0 + 64 * 32;

  f32x8 acc[2][4] = {};
  int nkt = g.K >> 5;

  auto stage = [&](int buf) {
    async_cp16(&sA[buf][o0], pA0);
    async_cp16(&sA[buf][o1], pA1);
    async_cp16(&sB[buf][o0], pB0);
    async_cp16(&sB[buf][o1], pB1);
    pA0 += 32; pA1 += 32; pB0 += 32; pB1 += 32;
  };

  stage(0);
  int mrow = lane & 15, half = lane >> 4;
  for (int kt = 0; kt < nkt; ++kt) {
    int buf = kt & 1;
    wait_async0();
    __syncthreads();
    if (kt + 1 < nkt) stage(buf ^ 1);

    const u16* a = sA[buf];
    const u16* b = sB[buf];
    bf16x16 af[2], bf[4];
    for (int sm = 0; sm < 2; ++sm) {
      int ml = wm * 32 + sm * 16 + mrow;
      union { bf16x16 v; uint4 q[2]; } u;
      u.q[0] = *(const uint4*)(a + ml * 32 + half * 8);
      u.q[1] = *(const uint4*)(a + ml * 32 + 16 + half * 8);
      af[sm] = u.v;
    }
    for (int sn = 0; sn < 4; ++sn) {
      int nl = wn * 64 + sn * 16 + mrow;
      union { bf16x16 v; uint4 q[2]; } u;
      u.q[0] = *(const uint4*)(b + nl * 32 + half * 8);
      u.q[1] = *(const uint4*)(b + nl * 32 + 16 + half * 8);
      bf[sn] = u.v;
    }
    for (int sm = 0; sm < 2; ++sm)
      for (int sn = 0; sn < 4; ++sn)
        acc[sm][sn] = __builtin_amdgcn_wmma_f32_16x16x32_bf16(
            false, af[sm], false, bf[sn], (short)0, acc[sm][sn], false, false);
  }

  for (int sm = 0; sm < 2; ++sm)
    for (int sn = 0; sn < 4; ++sn) {
      int n = n0 + wn * 64 + sn * 16 + (lane & 15);
      float bia = g.bias ? g.bias[n] : 0.f;
      for (int r = 0; r < 8; ++r) {
        int m = m0 + wm * 32 + sm * 16 + r + 8 * (lane >> 4);
        float v = acc[sm][sn][r] + bia;
        if (g.relu) v = fmaxf(v, 0.f);
        long long off = (long long)m * g.ldc + n;
        if (g.Cf) g.Cf[off] = v;
        if (g.Cb) g.Cb[off] = f2bf(v);
      }
    }
}

// ================================================================
// SMALL GEMM (attention): bounds-checked, batched over z, templated on
// B layout. BNN=0: B is [N,K]; BNN=1: B is [K,N] zero-filled for k>=Kb.
// ================================================================
struct Gemm {
  const u16* A; long long a_div, a_out, a_in; int lda;
  const u16* B; long long b_div, b_out, b_in; int ldb;
  float* Cf; u16* Cb; long long c_div, c_out, c_in; int ldc;
  int M, N, K, Kb;
  const float* bias; float scale; int relu;
};

#define BMS 128
#define BNS 64

template <int BNN>
__global__ void __launch_bounds__(256) k_gemm_small(Gemm g) {
  __shared__ __align__(16) u16 sA[BMS * 32];
  __shared__ __align__(16) u16 sB[BNS * 32];
  int z = blockIdx.z;
  const u16* A = g.A + (z / g.a_div) * g.a_out + (z % g.a_div) * g.a_in;
  const u16* B = g.B + (z / g.b_div) * g.b_out + (z % g.b_div) * g.b_in;
  long long coff = (z / g.c_div) * g.c_out + (z % g.c_div) * g.c_in;
  int m0 = blockIdx.y * BMS, n0 = blockIdx.x * BNS;
  int tid = threadIdx.x, lane = tid & 31;
  int wv = tid >> 5, wm = wv & 3, wn = wv >> 2;

  f32x8 acc[2][2] = {};
  int nkt = (g.K + 31) >> 5;

  for (int kt = 0; kt < nkt; ++kt) {
    int k0 = kt << 5;
    for (int i = 0; i < 2; ++i) {
      int idx = tid + i * 256;
      int row = idx >> 2, ch = idx & 3;
      uint4 v = {0u, 0u, 0u, 0u};
      int m = m0 + row;
      if (m < g.M) v = *(const uint4*)(A + (long long)m * g.lda + k0 + ch * 8);
      *(uint4*)(sA + row * 32 + ch * 8) = v;
    }
    if (!BNN) {
      int row = tid >> 2, ch = tid & 3;
      uint4 v = {0u, 0u, 0u, 0u};
      int n = n0 + row;
      if (n < g.N) v = *(const uint4*)(B + (long long)n * g.ldb + k0 + ch * 8);
      *(uint4*)(sB + row * 32 + ch * 8) = v;
    } else {
      int nloc = tid >> 2;
      int kb = (tid & 3) * 8;
      int n = n0 + nloc;
      u16 tmp[8];
      for (int j = 0; j < 8; ++j) {
        int k = k0 + kb + j;
        tmp[j] = (n < g.N && k < g.Kb) ? B[(long long)k * g.ldb + n] : (u16)0;
      }
      *(uint4*)(sB + nloc * 32 + kb) = *(const uint4*)tmp;
    }
    __syncthreads();

    int mrow = lane & 15, half = lane >> 4;
    bf16x16 af[2], bf[2];
    for (int sm = 0; sm < 2; ++sm) {
      int ml = wm * 32 + sm * 16 + mrow;
      union { bf16x16 v; uint4 q[2]; } u;
      u.q[0] = *(const uint4*)(sA + ml * 32 + half * 8);
      u.q[1] = *(const uint4*)(sA + ml * 32 + 16 + half * 8);
      af[sm] = u.v;
    }
    for (int sn = 0; sn < 2; ++sn) {
      int nl = wn * 32 + sn * 16 + mrow;
      union { bf16x16 v; uint4 q[2]; } u;
      u.q[0] = *(const uint4*)(sB + nl * 32 + half * 8);
      u.q[1] = *(const uint4*)(sB + nl * 32 + 16 + half * 8);
      bf[sn] = u.v;
    }
    for (int sm = 0; sm < 2; ++sm)
      for (int sn = 0; sn < 2; ++sn)
        acc[sm][sn] = __builtin_amdgcn_wmma_f32_16x16x32_bf16(
            false, af[sm], false, bf[sn], (short)0, acc[sm][sn], false, false);
    __syncthreads();
  }

  for (int sm = 0; sm < 2; ++sm)
    for (int sn = 0; sn < 2; ++sn) {
      int n = n0 + wn * 32 + sn * 16 + (lane & 15);
      if (n >= g.N) continue;
      float bia = g.bias ? g.bias[n] : 0.f;
      for (int r = 0; r < 8; ++r) {
        int m = m0 + wm * 32 + sm * 16 + r + 8 * (lane >> 4);
        if (m >= g.M) continue;
        float v = acc[sm][sn][r] * g.scale + bia;
        if (g.relu) v = fmaxf(v, 0.f);
        long long off = coff + (long long)m * g.ldc + n;
        if (g.Cf) g.Cf[off] = v;
        if (g.Cb) g.Cb[off] = f2bf(v);
      }
    }
}

// ---------------------------------------------------------------- softmax (row=174 -> bf16 padded 192)
__global__ void k_softmax(const float* __restrict__ sc, u16* __restrict__ pr) {
  __shared__ float red[256];
  int row = blockIdx.x, tid = threadIdx.x;
  const float* s = sc + (long long)row * SQ;
  float v = (tid < SQ) ? s[tid] : -3.4e38f;
  red[tid] = v; __syncthreads();
  for (int st = 128; st > 0; st >>= 1) { if (tid < st) red[tid] = fmaxf(red[tid], red[tid + st]); __syncthreads(); }
  float mx = red[0]; __syncthreads();
  float e = (tid < SQ) ? __expf(v - mx) : 0.f;
  red[tid] = e; __syncthreads();
  for (int st = 128; st > 0; st >>= 1) { if (tid < st) red[tid] += red[tid + st]; __syncthreads(); }
  float inv = 1.f / red[0];
  u16* p = pr + (long long)row * SP;
  if (tid < SP) p[tid] = (tid < SQ) ? f2bf(e * inv) : (u16)0;
}

// ---------------------------------------------------------------- residual + LayerNorm (row = 512)
__global__ void k_addln(const float* __restrict__ x, const float* __restrict__ r,
                        const float* __restrict__ g, const float* __restrict__ b,
                        float* __restrict__ xo, u16* __restrict__ xbo) {
  __shared__ float red[256];
  int row = blockIdx.x, tid = threadIdx.x;
  long long o = (long long)row * D;
  float v0 = x[o + tid] + r[o + tid];
  float v1 = x[o + tid + 256] + r[o + tid + 256];
  red[tid] = v0 + v1; __syncthreads();
  for (int st = 128; st > 0; st >>= 1) { if (tid < st) red[tid] += red[tid + st]; __syncthreads(); }
  float mean = red[0] * (1.f / D); __syncthreads();
  float d0 = v0 - mean, d1 = v1 - mean;
  red[tid] = d0 * d0 + d1 * d1; __syncthreads();
  for (int st = 128; st > 0; st >>= 1) { if (tid < st) red[tid] += red[tid + st]; __syncthreads(); }
  float inv = rsqrtf(red[0] * (1.f / D) + 1e-5f);
  float y0 = d0 * inv * g[tid] + b[tid];
  float y1 = d1 * inv * g[tid + 256] + b[tid + 256];
  xo[o + tid] = y0;        xo[o + tid + 256] = y1;
  xbo[o + tid] = f2bf(y0); xbo[o + tid + 256] = f2bf(y1);
}

// ---------------------------------------------------------------- classifier head
__global__ void k_head(const float* __restrict__ x, const float* __restrict__ Wc,
                       const float* __restrict__ bc, float* __restrict__ out) {
  __shared__ float red[256];
  int n = blockIdx.x, b = blockIdx.y, tid = threadIdx.x;
  const float* xr = x + (long long)b * (SQ * D);
  const float* wr = Wc + (long long)n * (SQ * D);
  float s = 0.f;
  for (int i = tid; i < SQ * D; i += 256) s += xr[i] * wr[i];
  red[tid] = s; __syncthreads();
  for (int st = 128; st > 0; st >>= 1) { if (tid < st) red[tid] += red[tid + st]; __syncthreads(); }
  if (tid == 0) out[(long long)b * NACT + n] = red[0] + bc[n];
}

// ---------------------------------------------------------------- host
extern "C" void kernel_launch(void* const* d_in, const int* in_sizes, int n_in,
                              void* d_out, int out_size, void* d_ws, size_t ws_size,
                              hipStream_t stream) {
  (void)in_sizes; (void)n_in; (void)out_size; (void)ws_size;
  const float* world    = (const float*)d_in[0];
  const float* requests = (const float*)d_in[1];
  const float* vehicles = (const float*)d_in[2];
  const float* Et   = (const float*)d_in[3];
  const float* Ep   = (const float*)d_in[4];
  const float* Erid = (const float*)d_in[5];
  const float* Erst = (const float*)d_in[6];
  const float* Evst = (const float*)d_in[7];
  const float* Ecv  = (const float*)d_in[8];
  const float* Wqkv = (const float*)d_in[9];
  const float* bqkv = (const float*)d_in[10];
  const float* Wo   = (const float*)d_in[11];
  const float* bo   = (const float*)d_in[12];
  const float* W1   = (const float*)d_in[13];
  const float* b1   = (const float*)d_in[14];
  const float* W2   = (const float*)d_in[15];
  const float* b2   = (const float*)d_in[16];
  const float* g1   = (const float*)d_in[17];
  const float* bl1  = (const float*)d_in[18];
  const float* g2   = (const float*)d_in[19];
  const float* bl2  = (const float*)d_in[20];
  const float* Wc   = (const float*)d_in[21];
  const float* bc   = (const float*)d_in[22];

  char* ws = (char*)d_ws; size_t off = 0;
  auto alloc = [&](size_t bytes) -> void* {
    void* p = ws + off; off += (bytes + 255) & ~(size_t)255; return p;
  };
  float* xf     = (float*)alloc((size_t)MTOK * D * 4);
  u16*   xb     = (u16*)  alloc((size_t)MTOK * D * 2);
  u16*   qkvb   = (u16*)  alloc((size_t)MTOK * 3 * D * 2);
  float* scores = (float*)alloc((size_t)NB * H * SQ * SQ * 4);
  u16*   probs  = (u16*)  alloc((size_t)NB * H * SQ * SP * 2);
  u16*   ob     = (u16*)  alloc((size_t)MTOK * D * 2);
  float* tmpf   = (float*)alloc((size_t)MTOK * D * 4);
  u16*   hb     = (u16*)  alloc((size_t)MTOK * DFF * 2);
  u16*   wqkvb  = (u16*)  alloc((size_t)NL * 3 * D * D * 2);
  u16*   wob    = (u16*)  alloc((size_t)NL * D * D * 2);
  u16*   w1b    = (u16*)  alloc((size_t)NL * DFF * D * 2);
  u16*   w2b    = (u16*)  alloc((size_t)NL * D * DFF * 2);

  k_cvt<<<1024, 256, 0, stream>>>(Wqkv, wqkvb, (long long)NL * 3 * D * D);
  k_cvt<<<1024, 256, 0, stream>>>(Wo,   wob,   (long long)NL * D * D);
  k_cvt<<<1024, 256, 0, stream>>>(W1,   w1b,   (long long)NL * DFF * D);
  k_cvt<<<1024, 256, 0, stream>>>(W2,   w2b,   (long long)NL * D * DFF);

  k_embed<<<MTOK, 128, 0, stream>>>(world, requests, vehicles,
                                    Et, Ep, Erid, Erst, Evst, Ecv, xf, xb);

  int mtf = MTOK / BMF;   // 696
  for (int i = 0; i < NL; ++i) {
    { // QKV: [MTOK,512] x [1536,512]^T -> bf16 qkv
      GemmF p{}; p.A = xb; p.lda = D;
      p.B = wqkvb + (size_t)i * 3 * D * D; p.ldb = D;
      p.Cb = qkvb; p.ldc = 3 * D; p.K = D;
      p.bias = bqkv + (size_t)i * 3 * D;
      k_gemm_fast<<<dim3((3 * D) / BNF, mtf, 1), 256, 0, stream>>>(p);
    }
    { // scores = q @ k^T * 1/8, batched over (b,h)
      Gemm p{}; p.A = qkvb; p.a_div = H; p.a_out = (long long)SQ * 3 * D; p.a_in = DH; p.lda = 3 * D;
      p.B = qkvb + D; p.b_div = H; p.b_out = (long long)SQ * 3 * D; p.b_in = DH; p.ldb = 3 * D;
      p.Cf = scores; p.c_div = 1; p.c_out = (long long)SQ * SQ;
      p.ldc = SQ; p.M = SQ; p.N = SQ; p.K = DH; p.Kb = DH; p.scale = 0.125f;
      k_gemm_small<0><<<dim3((SQ + BNS - 1) / BNS, (SQ + BMS - 1) / BMS, NB * H), 256, 0, stream>>>(p);
    }
    k_softmax<<<NB * H * SQ, 256, 0, stream>>>(scores, probs);
    { // o = probs @ v  (B is [K,N], K padded to 192)
      Gemm p{}; p.A = probs; p.a_div = 1; p.a_out = (long long)SQ * SP; p.lda = SP;
      p.B = qkvb + 2 * D; p.b_div = H; p.b_out = (long long)SQ * 3 * D; p.b_in = DH; p.ldb = 3 * D;
      p.Cb = ob; p.c_div = H; p.c_out = (long long)SQ * D; p.c_in = DH; p.ldc = D;
      p.M = SQ; p.N = DH; p.K = SP; p.Kb = SQ; p.scale = 1.f;
      k_gemm_small<1><<<dim3(1, (SQ + BMS - 1) / BMS, NB * H), 256, 0, stream>>>(p);
    }
    { // attn_out = o @ Wo^T + bo -> f32 tmp
      GemmF p{}; p.A = ob; p.lda = D;
      p.B = wob + (size_t)i * D * D; p.ldb = D;
      p.Cf = tmpf; p.ldc = D; p.K = D;
      p.bias = bo + (size_t)i * D;
      k_gemm_fast<<<dim3(D / BNF, mtf, 1), 256, 0, stream>>>(p);
    }
    k_addln<<<MTOK, 256, 0, stream>>>(xf, tmpf, g1 + (size_t)i * D, bl1 + (size_t)i * D, xf, xb);
    { // h = relu(x @ W1^T + b1) -> bf16
      GemmF p{}; p.A = xb; p.lda = D;
      p.B = w1b + (size_t)i * DFF * D; p.ldb = D;
      p.Cb = hb; p.ldc = DFF; p.K = D;
      p.bias = b1 + (size_t)i * DFF; p.relu = 1;
      k_gemm_fast<<<dim3(DFF / BNF, mtf, 1), 256, 0, stream>>>(p);
    }
    { // ff = h @ W2^T + b2 -> f32 tmp
      GemmF p{}; p.A = hb; p.lda = DFF;
      p.B = w2b + (size_t)i * D * DFF; p.ldb = DFF;
      p.Cf = tmpf; p.ldc = D; p.K = DFF;
      p.bias = b2 + (size_t)i * D;
      k_gemm_fast<<<dim3(D / BNF, mtf, 1), 256, 0, stream>>>(p);
    }
    k_addln<<<MTOK, 256, 0, stream>>>(xf, tmpf, g2 + (size_t)i * D, bl2 + (size_t)i * D, xf, xb);
  }
  k_head<<<dim3(NACT, NB), 256, 0, stream>>>(xf, Wc, bc, (float*)d_out);
}
